// STFTModel_4226247819476
// MI455X (gfx1250) — matrix operand order repaired and checked by reading, not compiled, not run on hardware
//
#include <hip/hip_runtime.h>

// ---------------------------------------------------------------------------
// STFT power spectrogram (dB) as an f16-WMMA GEMM for gfx1250 (MI455X).
//   M = frames (per batch), N = channels (cos block + sin block), K = 2048.
//   Each wave computes TWO 16x16 M-tiles for one N-tile (cos AND sin):
//   4 x v_wmma_f32_16x16x32_f16 per K-step on 8 b128 loads (2:1 ratio).
// ---------------------------------------------------------------------------

typedef __attribute__((ext_vector_type(16))) _Float16 v16h;
typedef __attribute__((ext_vector_type(8)))  _Float16 v8h;
typedef __attribute__((ext_vector_type(8)))  float    v8f;

#define WINDOW  2048
#define HOP     128
#define BATCH   16
#define TLEN    441000
#define TPAD    (TLEN + 2 * WINDOW)  // zero tail covers M padded to 3456 frames
#define FRAMES  3430                 // (TLEN - WINDOW)/HOP + 1
#define NCH     1025
#define NCHP    1040                 // channels padded to multiple of 16
#define MTILES  215                  // ceil(FRAMES / 16)
#define MPAIRS  108                  // ceil(MTILES / 2) -> 3456 frames padded
#define CTILES  (NCHP / 16)          // 65
#define WAVES_PER_BLOCK 8

// ---- f32 -> f16 conversion of the signal, padded per batch ---------------
__global__ __launch_bounds__(256)
void cvt_x_kernel(const float* __restrict__ x, _Float16* __restrict__ xh) {
    size_t i = (size_t)blockIdx.x * 256 + threadIdx.x;
    if (i >= (size_t)BATCH * TPAD) return;
    int b = (int)(i / TPAD);
    int t = (int)(i % TPAD);
    float v = (t < TLEN) ? x[(size_t)b * TLEN + t] : 0.0f;
    xh[i] = (_Float16)v;
}

// ---- f32 -> f16 conversion of cos/sin weights, rows padded to NCHP -------
__global__ __launch_bounds__(256)
void cvt_w_kernel(const float* __restrict__ cw, const float* __restrict__ sw,
                  _Float16* __restrict__ wh) {
    size_t i = (size_t)blockIdx.x * 256 + threadIdx.x;
    if (i >= (size_t)2 * NCHP * WINDOW) return;
    int trig = (int)(i / ((size_t)NCHP * WINDOW));
    int rem  = (int)(i % ((size_t)NCHP * WINDOW));
    int row  = rem / WINDOW;
    int k    = rem % WINDOW;
    const float* src = trig ? sw : cw;
    float v = (row < NCH) ? src[(size_t)row * WINDOW + k] : 0.0f;
    wh[i] = (_Float16)v;
}

static __device__ __forceinline__ v16h cat16(v8h lo, v8h hi) {
    return __builtin_shufflevector(lo, hi, 0, 1, 2, 3, 4, 5, 6, 7,
                                           8, 9, 10, 11, 12, 13, 14, 15);
}

// ---- main GEMM: each wave -> 2 M-tiles x 1 N-tile, cos+sin accumulators ---
__global__ __launch_bounds__(32 * WAVES_PER_BLOCK)
void stft_wmma_kernel(const _Float16* __restrict__ xh,
                      const _Float16* __restrict__ wh,
                      float* __restrict__ out) {
    const int wave  = threadIdx.x >> 5;
    const int lane  = threadIdx.x & 31;
    const int mpair = blockIdx.x * WAVES_PER_BLOCK + wave;
    if (mpair >= MPAIRS) return;                 // wave-uniform: EXEC stays full
    const int mtile0 = mpair * 2;                // second tile = mtile0 + 1
    const int ctile  = blockIdx.y;
    const int b      = blockIdx.z;

    const int half  = lane >> 4;                 // lane half selects K sub-block
    const int idx16 = lane & 15;                 // A: row (frame) / B: col (chan)

    // A: frame rows are contiguous in time -> contiguous in K.
    // Tile mtile0+1 sits at a constant +16*HOP elements (folds into ioffset).
    const _Float16* aBase = xh + (size_t)b * TPAD
                               + (size_t)(mtile0 * 16 + idx16) * HOP;
    // B: weight rows [channel][K] row-major -> contiguous in K.
    const _Float16* cBase = wh + (size_t)(ctile * 16 + idx16) * WINDOW;
    const _Float16* sBase = cBase + (size_t)NCHP * WINDOW;

    v8f accC0 = {}, accS0 = {};
    v8f accC1 = {}, accS1 = {};

#pragma unroll 2
    for (int k0 = 0; k0 < WINDOW; k0 += 32) {
        // 16-bit A layout: half 0 -> K {0..7, 16..23}; half 1 -> K {8..15, 24..31}
        v16h a0 = cat16(*(const v8h*)(aBase + k0 + half * 8),
                        *(const v8h*)(aBase + k0 + 16 + half * 8));
        v16h a1 = cat16(*(const v8h*)(aBase + 16 * HOP + k0 + half * 8),
                        *(const v8h*)(aBase + 16 * HOP + k0 + 16 + half * 8));

        // 16-bit B layout: half 0 -> K {0..7, 16..23}; half 1 -> K {8..15, 24..31}
        v16h bc = cat16(*(const v8h*)(cBase + k0 + half * 8),
                        *(const v8h*)(cBase + k0 + 16 + half * 8));
        v16h bs = cat16(*(const v8h*)(sBase + k0 + half * 8),
                        *(const v8h*)(sBase + k0 + 16 + half * 8));

        accC0 = __builtin_amdgcn_wmma_f32_16x16x32_f16(
            false, a0, false, bc, (short)0, accC0, false, false);
        accS0 = __builtin_amdgcn_wmma_f32_16x16x32_f16(
            false, a0, false, bs, (short)0, accS0, false, false);
        accC1 = __builtin_amdgcn_wmma_f32_16x16x32_f16(
            false, a1, false, bc, (short)0, accC1, false, false);
        accS1 = __builtin_amdgcn_wmma_f32_16x16x32_f16(
            false, a1, false, bs, (short)0, accS1, false, false);
    }

    // C/D layout: VGPR r, lanes 0-15 -> M=r, lanes 16-31 -> M=r+8; lane&15 -> N.
    const int ch = ctile * 16 + idx16;
    if (ch < NCH) {
        const size_t outRow = ((size_t)b * NCH + ch) * FRAMES;
#pragma unroll
        for (int r = 0; r < 8; ++r) {
            int f0 = mtile0 * 16 + r + half * 8;
            if (f0 < FRAMES) {
                float c = accC0[r], s = accS0[r];
                out[outRow + f0] = 10.0f * __log10f(c * c + s * s);
            }
            int f1 = f0 + 16;
            if (f1 < FRAMES) {
                float c = accC1[r], s = accS1[r];
                out[outRow + f1] = 10.0f * __log10f(c * c + s * s);
            }
        }
    }
}

extern "C" void kernel_launch(void* const* d_in, const int* in_sizes, int n_in,
                              void* d_out, int out_size, void* d_ws, size_t ws_size,
                              hipStream_t stream) {
    const float* x  = (const float*)d_in[0];   // [16, 1, 441000] f32
    const float* cw = (const float*)d_in[1];   // [1025, 1, 2048] f32
    const float* sw = (const float*)d_in[2];   // [1025, 1, 2048] f32
    float* out = (float*)d_out;                // [16, 1025, 3430] f32

    // Workspace: f16 signal (padded) then f16 weights (cos block, sin block).
    _Float16* xh = (_Float16*)d_ws;
    _Float16* wh = (_Float16*)((char*)d_ws +
                               (size_t)BATCH * TPAD * sizeof(_Float16));

    const int n1 = BATCH * TPAD;
    cvt_x_kernel<<<(n1 + 255) / 256, 256, 0, stream>>>(x, xh);

    const int n2 = 2 * NCHP * WINDOW;
    cvt_w_kernel<<<(n2 + 255) / 256, 256, 0, stream>>>(cw, sw, wh);

    dim3 grid((MPAIRS + WAVES_PER_BLOCK - 1) / WAVES_PER_BLOCK,  // 14
              CTILES,                                            // 65
              BATCH);                                            // 16
    stft_wmma_kernel<<<grid, 32 * WAVES_PER_BLOCK, 0, stream>>>(xh, wh, out);
}
